// PhysicsInformedCMBNet_66168266162954
// MI455X (gfx1250) — compile-verified, matrix operand order checked by reading, not run on hardware
//
#include <hip/hip_runtime.h>
#include <hip/hip_bf16.h>
#include <math.h>

typedef __attribute__((ext_vector_type(16))) _Float16 v16h;
typedef __attribute__((ext_vector_type(8)))  _Float16 v8h;
typedef __attribute__((ext_vector_type(4)))  _Float16 v4h;
typedef __attribute__((ext_vector_type(8)))  float    v8f;
typedef __attribute__((ext_vector_type(4)))  float    v4f;
typedef __attribute__((ext_vector_type(4)))  unsigned v4u;
typedef __attribute__((ext_vector_type(8)))  int      v8i;
typedef __attribute__((ext_vector_type(4)))  int      v4i;

#if __has_builtin(__builtin_amdgcn_tensor_load_to_lds) && \
    __has_builtin(__builtin_amdgcn_s_wait_tensorcnt)
#define HAVE_TDM 1
#else
#define HAVE_TDM 0
#endif

__device__ __forceinline__ v16h cat16(v8h lo, v8h hi) {
  return __builtin_shufflevector(lo, hi, 0,1,2,3,4,5,6,7,8,9,10,11,12,13,14,15);
}

// ---------------------------------------------------------------------------
// Conv as implicit GEMM on v_wmma_f32_16x16x32_f16 (wave32), f16 NHWC in,
// f32 NHWC out.  2x2 register tiling per wave: 32 out-channels x 32 pixels,
// 4 WMMAs per 32-wide k-step sharing 2 A-fragments (LDS) + 2 B-fragments.
// Weights pre-packed f16 [Cout][KHW*Cin] (khw-major, padded to mult of 32),
// staged per-block into LDS by the TDM as a 2-D tile (32 rows x KTOTP elems,
// data_size=2B), waited with s_wait_tensorcnt, cooperative-copy fallback.
// Fragment index map per ISA 7.12.2: halves 0-7 -> k = kk + (lane>>4)*8 + e,
// halves 8-15 -> k = kk + 16 + (lane>>4)*8 + (e-8).
// ---------------------------------------------------------------------------
template<int Cin,int Cout,int H,int W,int KH,int KW,int PAD>
__global__ __launch_bounds__(128)
void conv_wmma(const _Float16* __restrict__ in, const _Float16* __restrict__ wpk,
               const float* __restrict__ bias, float* __restrict__ out) {
  constexpr int KHW   = KH * KW;
  constexpr int KTOT  = Cin * KHW;
  constexpr int KSTEP = (KTOT + 31) / 32;
  constexpr int KTOTP = KSTEP * 32;
  __shared__ _Float16 smemW[32 * KTOTP];

  const int lane   = threadIdx.x & 31;
  const int wave   = threadIdx.x >> 5;
  const int ocBase = blockIdx.y * 32;
  const _Float16* __restrict__ wsrc = wpk + (size_t)ocBase * KTOTP;

  if (wave == 0) {
#if HAVE_TDM
    // D# group0/group1 per CDNA5 ISA 8.3/8.4: count=1, type=2 (image),
    // data_size=1 (2 bytes), 2-D tile: 32 rows x KTOTP elems, stride KTOTP.
    const unsigned long long ga = (unsigned long long)(const void*)wsrc;
    const unsigned lds  = (unsigned)(unsigned long long)(const void*)&smemW[0];
    const unsigned kt   = (unsigned)KTOTP;              // <= 2304, fits 16 bits
    const v4u g0 = { 1u,                                 // count=1
                     lds,                                // lds_addr
                     (unsigned)ga,                       // global_addr[31:0]
                     (unsigned)((ga >> 32) & 0x01FFFFFFu) | (2u << 30) };
    const v8i g1 = { (int)(1u << 16),                    // data_size = 2B
                     (int)((kt & 0xFFFFu) << 16),        // tensor_dim0[15:0]
                     (int)((kt >> 16) | (32u << 16)),    // dim0[31:16], dim1=32
                     (int)((kt & 0xFFFFu) << 16),        // tile_dim0 = KTOTP
                     (int)32,                            // tile_dim1=32, dim2=0
                     (int)kt,                            // dim0_stride[31:0]
                     0, 0 };
    const v4i gz = {0, 0, 0, 0};
#if defined(__clang_major__) && __clang_major__ >= 23
    const v8i gz8 = {0, 0, 0, 0, 0, 0, 0, 0};
    __builtin_amdgcn_tensor_load_to_lds(g0, g1, gz, gz, gz8, 0);
#else
    __builtin_amdgcn_tensor_load_to_lds(g0, g1, gz, gz, 0);
#endif
    __builtin_amdgcn_s_wait_tensorcnt(0);
#else
    for (int i = lane; i < 32 * KTOTP; i += 32) smemW[i] = wsrc[i];
#endif
  }
  __syncthreads();

  const int m  = lane & 15;
  const int hi = lane >> 4;
  const int pixBase = (blockIdx.x * 4 + wave) * 32;   // 32 consecutive pixels,
  const int bimg = pixBase / (H * W);                 // row-aligned (W%32==0)
  const int rem  = pixBase % (H * W);
  const int oh  = rem / W;
  const int owB = rem % W;
  const int ow0 = owB + m;
  const int ow1 = ow0 + 16;
  const _Float16* __restrict__ arow0 = smemW + m        * KTOTP + hi * 8;
  const _Float16* __restrict__ arow1 = smemW + (16 + m) * KTOTP + hi * 8;

  v8f acc00 = {}, acc01 = {}, acc10 = {}, acc11 = {};
  if constexpr ((Cin & 31) == 0) {
    for (int ks = 0; ks < KSTEP; ++ks) {
      const int kk  = ks * 32;
      const int khw = kk / Cin;            // constant over the 32-k chunk
      const int ci0 = kk - khw * Cin;
      const int kh  = khw / KW;
      const int kw  = khw - kh * KW;
      const int iy  = oh + kh - PAD;       // wave-uniform
      const int ix0 = ow0 + kw - PAD;      // per-lane
      const int ix1 = ix0 + 16;
      v8h b0lo = {}, b0hi = {}, b1lo = {}, b1hi = {};
      if (iy >= 0 && iy < H) {
        const size_t rowoff = (size_t)(bimg * H + iy) * W;
        if (ix0 >= 0 && ix0 < W) {
          const _Float16* __restrict__ bp =
              in + (rowoff + ix0) * Cin + ci0 + hi * 8;
          b0lo = *(const v8h*)bp;
          b0hi = *(const v8h*)(bp + 16);
        }
        if (ix1 >= 0 && ix1 < W) {
          const _Float16* __restrict__ bp =
              in + (rowoff + ix1) * Cin + ci0 + hi * 8;
          b1lo = *(const v8h*)bp;
          b1hi = *(const v8h*)(bp + 16);
        }
      }
      const v16h a0 = cat16(*(const v8h*)(arow0 + kk), *(const v8h*)(arow0 + kk + 16));
      const v16h a1 = cat16(*(const v8h*)(arow1 + kk), *(const v8h*)(arow1 + kk + 16));
      const v16h b0 = cat16(b0lo, b0hi);
      const v16h b1 = cat16(b1lo, b1hi);
      acc00 = __builtin_amdgcn_wmma_f32_16x16x32_f16(false, a0, false, b0,
                                                     (short)0, acc00, false, false);
      acc01 = __builtin_amdgcn_wmma_f32_16x16x32_f16(false, a0, false, b1,
                                                     (short)0, acc01, false, false);
      acc10 = __builtin_amdgcn_wmma_f32_16x16x32_f16(false, a1, false, b0,
                                                     (short)0, acc10, false, false);
      acc11 = __builtin_amdgcn_wmma_f32_16x16x32_f16(false, a1, false, b1,
                                                     (short)0, acc11, false, false);
    }
  } else {
    // scalar-gather path (conv1: Cin==1, KTOT==25, single padded k-step)
    for (int ks = 0; ks < KSTEP; ++ks) {
      const int k0 = ks * 32;
      v16h a0, a1, b0, b1;
#pragma unroll
      for (int e = 0; e < 16; ++e) {
        const int k  = k0 + ((e < 8) ? e : (8 + e));  // + hi*8 via arow offset
        const int kg = k + hi * 8;                    // true reduction index
        a0[e] = arow0[k];                             // zero-padded weights
        a1[e] = arow1[k];
        _Float16 bv0 = (_Float16)0.f, bv1 = (_Float16)0.f;
        if (kg < KTOT) {
          const int kh = kg / KW;                     // Cin==1 -> k == khw
          const int kw = kg - kh * KW;
          const int iy = oh + kh - PAD;
          if (iy >= 0 && iy < H) {
            const size_t rowoff = (size_t)(bimg * H + iy) * W;
            const int ix0 = ow0 + kw - PAD;
            const int ix1 = ix0 + 16;
            if (ix0 >= 0 && ix0 < W) bv0 = in[rowoff + ix0];
            if (ix1 >= 0 && ix1 < W) bv1 = in[rowoff + ix1];
          }
        }
        b0[e] = bv0;
        b1[e] = bv1;
      }
      acc00 = __builtin_amdgcn_wmma_f32_16x16x32_f16(false, a0, false, b0,
                                                     (short)0, acc00, false, false);
      acc01 = __builtin_amdgcn_wmma_f32_16x16x32_f16(false, a0, false, b1,
                                                     (short)0, acc01, false, false);
      acc10 = __builtin_amdgcn_wmma_f32_16x16x32_f16(false, a1, false, b0,
                                                     (short)0, acc10, false, false);
      acc11 = __builtin_amdgcn_wmma_f32_16x16x32_f16(false, a1, false, b1,
                                                     (short)0, acc11, false, false);
    }
  }
  // D: VGPR r -> oc = ocTile + 8*hi + r (8 consecutive), col = pixel. NHWC.
  const float* __restrict__ bp0 = bias + ocBase + 8 * hi;
  const float* __restrict__ bp1 = bp0 + 16;
  float* __restrict__ o00 = out + (size_t)(pixBase + m)      * Cout + ocBase + 8 * hi;
  float* __restrict__ o01 = out + (size_t)(pixBase + 16 + m) * Cout + ocBase + 8 * hi;
  float* __restrict__ o10 = o00 + 16;
  float* __restrict__ o11 = o01 + 16;
#pragma unroll
  for (int r = 0; r < 8; ++r) {
    o00[r] = acc00[r] + bp0[r];
    o01[r] = acc01[r] + bp0[r];
    o10[r] = acc10[r] + bp1[r];
    o11[r] = acc11[r] + bp1[r];
  }
}

// ---------------------------------------------------------------------------
// FC on WMMA: batch 16 == M. K % 32 == 0 for every layer -> fully vectorized
// f32 loads + convertvector.  Weight row clamped for the N-tail (junk columns
// are never stored).
// ---------------------------------------------------------------------------
__device__ __forceinline__ v16h frag_from_f32(const float* __restrict__ base) {
  const v4f f0 = *(const v4f*)(base);
  const v4f f1 = *(const v4f*)(base + 4);
  const v4f f2 = *(const v4f*)(base + 16);
  const v4f f3 = *(const v4f*)(base + 20);
  const v4h h0 = __builtin_convertvector(f0, v4h);
  const v4h h1 = __builtin_convertvector(f1, v4h);
  const v4h h2 = __builtin_convertvector(f2, v4h);
  const v4h h3 = __builtin_convertvector(f3, v4h);
  const v8h lo  = __builtin_shufflevector(h0, h1, 0,1,2,3,4,5,6,7);
  const v8h hi8 = __builtin_shufflevector(h2, h3, 0,1,2,3,4,5,6,7);
  return cat16(lo, hi8);
}

template<bool RELU>
__global__ __launch_bounds__(128)
void fc_wmma(const float* __restrict__ in, const float* __restrict__ wgt,
             const float* __restrict__ bias, float* __restrict__ out,
             int K, int Nout, int ldin, int ldout) {
  const int lane  = threadIdx.x & 31;
  const int wave  = threadIdx.x >> 5;
  const int nBase = (blockIdx.x * 4 + wave) * 16;
  if (nBase >= Nout) return;                 // wave-uniform
  const int m  = lane & 15;
  const int hi = lane >> 4;
  const int o  = nBase + m;
  const int oc = (o < Nout) ? o : (Nout - 1);
  const float* __restrict__ irow = in  + (size_t)m  * ldin + hi * 8;
  const float* __restrict__ wrow = wgt + (size_t)oc * K    + hi * 8;
  v8f acc = {};
  for (int kk = 0; kk < K; kk += 32) {
    const v16h a = frag_from_f32(irow + kk);
    const v16h b = frag_from_f32(wrow + kk);
    acc = __builtin_amdgcn_wmma_f32_16x16x32_f16(false, a, false, b,
                                                 (short)0, acc, false, false);
  }
  if (o < Nout) {
    const float bi = bias[o];
#pragma unroll
    for (int r = 0; r < 8; ++r) {
      float v = acc[r] + bi;
      if (RELU) v = fmaxf(v, 0.f);
      out[(size_t)(r + 8 * hi) * ldout + o] = v;
    }
  }
}

// ---------------------------------------------------------------------------
// Layout / precision helpers.
// ---------------------------------------------------------------------------
__global__ void f32_to_f16(const float* __restrict__ in, _Float16* __restrict__ out, int n) {
  const int i = blockIdx.x * blockDim.x + threadIdx.x;
  if (i < n) out[i] = (_Float16)in[i];
}

// OIHW f32 -> [Cout][KHW*Cin] f16, khw-major / ci-minor, zero-padded to KTOTP.
__global__ void wgt_pack(const float* __restrict__ w, _Float16* __restrict__ wp,
                         int Cout, int Cin, int KHW, int KTOTP) {
  const int idx = blockIdx.x * blockDim.x + threadIdx.x;
  if (idx >= Cout * KTOTP) return;
  const int kp = idx % KTOTP;
  const int oc = idx / KTOTP;
  const int KTOT = Cin * KHW;
  float v = 0.f;
  if (kp < KTOT) {
    const int khw = kp / Cin;
    const int ci  = kp - khw * Cin;
    v = w[(size_t)(oc * Cin + ci) * KHW + khw];
  }
  wp[idx] = (_Float16)v;
}

// ---------------------------------------------------------------------------
// BatchNorm (training-mode batch statistics) on f32 NHWC.
// ---------------------------------------------------------------------------
__global__ void bn_stats(const float* __restrict__ x, float* __restrict__ mean,
                         float* __restrict__ rstd, int C, int NPIX) {
  const int c = blockIdx.x;
  float s = 0.f, s2 = 0.f;
  for (int i = threadIdx.x; i < NPIX; i += 256) {
    const float v = x[(size_t)i * C + c];
    s += v; s2 += v * v;
  }
  __shared__ float sh[256], sh2[256];
  sh[threadIdx.x] = s; sh2[threadIdx.x] = s2;
  __syncthreads();
  for (int o = 128; o > 0; o >>= 1) {
    if (threadIdx.x < o) {
      sh[threadIdx.x]  += sh[threadIdx.x + o];
      sh2[threadIdx.x] += sh2[threadIdx.x + o];
    }
    __syncthreads();
  }
  if (threadIdx.x == 0) {
    const float n  = (float)NPIX;
    const float mu = sh[0] / n;
    const float va = sh2[0] / n - mu * mu;
    mean[c] = mu;
    rstd[c] = rsqrtf(va + 1e-5f);
  }
}

__global__ void bn_relu_pool2(const float* __restrict__ x,
                              const float* __restrict__ mean,
                              const float* __restrict__ rstd,
                              const float* __restrict__ g,
                              const float* __restrict__ bb,
                              _Float16* __restrict__ out, int C, int H, int W) {
  const int Ho = H >> 1, Wo = W >> 1;
  const int idx = blockIdx.x * blockDim.x + threadIdx.x;
  if (idx >= 16 * C * Ho * Wo) return;
  const int c = idx % C;
  int t = idx / C;
  const int ow = t % Wo; t /= Wo;
  const int oh = t % Ho;
  const int b  = t / Ho;
  const float sc = g[c] * rstd[c];
  const float sh = bb[c] - mean[c] * sc;
  const float* __restrict__ p =
      x + (size_t)((b * H + oh * 2) * W + ow * 2) * C + c;
  const float v0 = fmaxf(p[0] * sc + sh, 0.f);
  const float v1 = fmaxf(p[C] * sc + sh, 0.f);
  const float v2 = fmaxf(p[(size_t)W * C] * sc + sh, 0.f);
  const float v3 = fmaxf(p[(size_t)W * C + C] * sc + sh, 0.f);
  out[idx] = (_Float16)fmaxf(fmaxf(v0, v1), fmaxf(v2, v3));
}

__global__ void bn_relu(const float* __restrict__ x,
                        const float* __restrict__ mean,
                        const float* __restrict__ rstd,
                        const float* __restrict__ g,
                        const float* __restrict__ bb,
                        _Float16* __restrict__ out, int C, int total) {
  const int idx = blockIdx.x * blockDim.x + threadIdx.x;
  if (idx >= total) return;
  const int c = idx % C;
  out[idx] = (_Float16)fmaxf(g[c] * (x[idx] - mean[c]) * rstd[c] + bb[c], 0.f);
}

// adaptive avg-pool (4,4) over 32x32 NHWC f16 -> concat rows (stride 2112),
// channel-major flatten: cat[b, c*16 + ph*4 + pw].
__global__ void avgpool8_cat(const _Float16* __restrict__ x, float* __restrict__ cat) {
  const int idx = blockIdx.x * blockDim.x + threadIdx.x;  // 16*128*16
  if (idx >= 16 * 128 * 16) return;
  const int pw = idx & 3;
  int t = idx >> 2;
  const int ph = t & 3; t >>= 2;
  const int c  = t & 127;
  const int b  = t >> 7;
  float s = 0.f;
#pragma unroll
  for (int i = 0; i < 8; ++i)
#pragma unroll
    for (int j = 0; j < 8; ++j)
      s += (float)x[(size_t)((b * 32 + ph * 8 + i) * 32 + pw * 8 + j) * 128 + c];
  cat[b * 2112 + c * 16 + ph * 4 + pw] = s * (1.0f / 64.0f);
}

// ---------------------------------------------------------------------------
// 256-point radix-2 FFT (DIT) in LDS; rows pass then cols pass. f32.
// ---------------------------------------------------------------------------
__device__ __forceinline__ int bitrev8(int v) {
  v = ((v & 0x0F) << 4) | ((v >> 4) & 0x0F);
  v = ((v & 0x33) << 2) | ((v >> 2) & 0x33);
  v = ((v & 0x55) << 1) | ((v >> 1) & 0x55);
  return v;
}

__device__ __forceinline__ void fft256_stages(float* sr, float* si, int t) {
#pragma unroll
  for (int len = 2; len <= 256; len <<= 1) {
    const int half = len >> 1;
    const int grp  = t / half;
    const int off  = t - grp * half;
    const int pos  = grp * len + off;
    const float ang = -6.283185307179586f *
                      (float)(off * (256 / len)) * (1.0f / 256.0f);
    float s, c;
    __sincosf(ang, &s, &c);
    const float ur = sr[pos],        ui = si[pos];
    const float vr = sr[pos + half], vi = si[pos + half];
    const float tr = vr * c - vi * s;
    const float ti = vr * s + vi * c;
    sr[pos]        = ur + tr;  si[pos]        = ui + ti;
    sr[pos + half] = ur - tr;  si[pos + half] = ui - ti;
    __syncthreads();
  }
}

__global__ void fft256_rows(const float* __restrict__ x,
                            float* __restrict__ re, float* __restrict__ im) {
  __shared__ float sr[256], si[256];
  const int row = blockIdx.x;                // b*256 + y
  const float* __restrict__ src = x + (size_t)row * 256;
  const int t = threadIdx.x;                 // 128 threads
  for (int e = t; e < 256; e += 128) { sr[bitrev8(e)] = src[e]; si[bitrev8(e)] = 0.f; }
  __syncthreads();
  fft256_stages(sr, si, t);
  for (int e = t; e < 256; e += 128) {
    re[(size_t)row * 256 + e] = sr[e];
    im[(size_t)row * 256 + e] = si[e];
  }
}

__global__ void fft256_cols(float* __restrict__ re, float* __restrict__ im) {
  __shared__ float sr[256], si[256];
  const int gc = blockIdx.x;                 // b*256 + col
  const int b = gc >> 8, col = gc & 255;
  float* __restrict__ pr = re + (size_t)b * 65536 + col;
  float* __restrict__ pi = im + (size_t)b * 65536 + col;
  const int t = threadIdx.x;
  for (int e = t; e < 256; e += 128) {
    const int rv = bitrev8(e);
    sr[rv] = pr[(size_t)e * 256];
    si[rv] = pi[(size_t)e * 256];
  }
  __syncthreads();
  fft256_stages(sr, si, t);
  for (int e = t; e < 256; e += 128) {
    pr[(size_t)e * 256] = sr[e];
    pi[(size_t)e * 256] = si[e];
  }
}

// bin = floor(r/4), valid iff r < 128; deterministic (one thread per bin).
__global__ void power_radial(const float* __restrict__ re,
                             const float* __restrict__ im,
                             float* __restrict__ pf0) {
  const int b = blockIdx.x;
  const int bin = threadIdx.x;               // 32 threads = 32 bins
  const float* __restrict__ pr = re + (size_t)b * 65536;
  const float* __restrict__ pi = im + (size_t)b * 65536;
  float sum = 0.f;
  int cnt = 0;
  for (int p = 0; p < 65536; ++p) {
    const int y = p >> 8, x = p & 255;
    const float dy = (float)(y - 128), dx = (float)(x - 128);
    const float r = sqrtf(dx * dx + dy * dy);
    if (r < 128.0f && (int)floorf(r * 0.25f) == bin) {
      const float vr = pr[p], vi = pi[p];
      sum += vr * vr + vi * vi;
      ++cnt;
    }
  }
  const float radial = (cnt > 0) ? sum / (float)cnt : 0.f;
  pf0[b * 32 + bin] = logf(radial + 1e-10f);
}

// ---------------------------------------------------------------------------
extern "C" void kernel_launch(void* const* d_in, const int* in_sizes, int n_in,
                              void* d_out, int out_size, void* d_ws, size_t ws_size,
                              hipStream_t stream) {
  (void)in_sizes; (void)n_in; (void)out_size; (void)ws_size;
  const float* x   = (const float*)d_in[0];
  const float* cw1 = (const float*)d_in[1];
  const float* cb1 = (const float*)d_in[2];
  const float* g1  = (const float*)d_in[3];
  const float* bb1 = (const float*)d_in[4];
  const float* cw2 = (const float*)d_in[5];
  const float* cb2 = (const float*)d_in[6];
  const float* g2  = (const float*)d_in[7];
  const float* bb2 = (const float*)d_in[8];
  const float* cw3 = (const float*)d_in[9];
  const float* cb3 = (const float*)d_in[10];
  const float* g3  = (const float*)d_in[11];
  const float* bb3 = (const float*)d_in[12];
  const float* cw4 = (const float*)d_in[13];
  const float* cb4 = (const float*)d_in[14];
  const float* g4  = (const float*)d_in[15];
  const float* bb4 = (const float*)d_in[16];
  const float* pcw = (const float*)d_in[17];
  const float* pcb = (const float*)d_in[18];
  const float* pg  = (const float*)d_in[19];
  const float* pbb = (const float*)d_in[20];
  const float* pw1 = (const float*)d_in[21];
  const float* pb1 = (const float*)d_in[22];
  const float* pw2 = (const float*)d_in[23];
  const float* pb2 = (const float*)d_in[24];
  const float* pw3 = (const float*)d_in[25];
  const float* pb3 = (const float*)d_in[26];
  const float* fw1 = (const float*)d_in[27];
  const float* fb1 = (const float*)d_in[28];
  const float* fw2 = (const float*)d_in[29];
  const float* fb2 = (const float*)d_in[30];
  const float* fw3 = (const float*)d_in[31];
  const float* fb3 = (const float*)d_in[32];
  const float* fw4 = (const float*)d_in[33];
  const float* fb4 = (const float*)d_in[34];

  float* bump = (float*)d_ws;
  auto alloc = [&](size_t nfloats) { float* r = bump; bump += nfloats; return r; };
  float*     RAW = alloc(33554432);              // max raw conv out, f32 NHWC
  _Float16*  PAh = (_Float16*)alloc(4194304);    // 8,388,608 halves (pool ping)
  _Float16*  PBh = (_Float16*)alloc(2097152);    // 4,194,304 halves (pool pong)
  _Float16*  H4h = (_Float16*)alloc(2097152);    // [16,32,32,256] f16
  _Float16*  P5h = (_Float16*)alloc(1048576);    // [16,32,32,128] f16
  _Float16*  XH  = (_Float16*)alloc(524288);     // x as f16, 1,048,576 halves
  _Float16*  WP1 = (_Float16*)alloc(512);        // 32 x 32
  _Float16*  WP2 = (_Float16*)alloc(25600);      // 64 x 800
  _Float16*  WP3 = (_Float16*)alloc(36864);      // 128 x 576
  _Float16*  WP4 = (_Float16*)alloc(147456);     // 256 x 1152
  _Float16*  WP5 = (_Float16*)alloc(147456);     // 128 x 2304
  float*     FRE = alloc(1048576);
  float*     FIM = alloc(1048576);
  float*     PF0 = alloc(512);
  float*     PF1 = alloc(4096);
  float*     PF2 = alloc(2048);
  float*     CAT = alloc(33792);                 // [16,2112]
  float*     FC1 = alloc(8192);
  float*     FC2 = alloc(4096);
  float*     FC3 = alloc(2048);
  float*     MEAN = alloc(256);
  float*     RSTD = alloc(256);

  // --- precision / layout prep ---
  f32_to_f16<<<4096, 256, 0, stream>>>(x, XH, 1048576);
  wgt_pack<<<4,    256, 0, stream>>>(cw1, WP1,  32,   1, 25,   32);
  wgt_pack<<<200,  256, 0, stream>>>(cw2, WP2,  64,  32, 25,  800);
  wgt_pack<<<288,  256, 0, stream>>>(cw3, WP3, 128,  64,  9,  576);
  wgt_pack<<<1152, 256, 0, stream>>>(cw4, WP4, 256, 128,  9, 1152);
  wgt_pack<<<1152, 256, 0, stream>>>(pcw, WP5, 128, 256,  9, 2304);

  // --- conv1 [16,256,256,1] -> f32 NHWC [16,256,256,32] -> bn+relu+pool ---
  conv_wmma<1,32,256,256,5,5,2><<<dim3(8192, 1), 128, 0, stream>>>(XH, WP1, cb1, RAW);
  bn_stats<<<32, 256, 0, stream>>>(RAW, MEAN, RSTD, 32, 1048576);
  bn_relu_pool2<<<32768, 256, 0, stream>>>(RAW, MEAN, RSTD, g1, bb1, PAh, 32, 256, 256);

  // --- conv2 -> [16,128,128,64] -> pool -> PBh [16,64,64,64] ---
  conv_wmma<32,64,128,128,5,5,2><<<dim3(2048, 2), 128, 0, stream>>>(PAh, WP2, cb2, RAW);
  bn_stats<<<64, 256, 0, stream>>>(RAW, MEAN, RSTD, 64, 262144);
  bn_relu_pool2<<<16384, 256, 0, stream>>>(RAW, MEAN, RSTD, g2, bb2, PBh, 64, 128, 128);

  // --- conv3 -> [16,64,64,128] -> pool -> PAh [16,32,32,128] ---
  conv_wmma<64,128,64,64,3,3,1><<<dim3(512, 4), 128, 0, stream>>>(PBh, WP3, cb3, RAW);
  bn_stats<<<128, 256, 0, stream>>>(RAW, MEAN, RSTD, 128, 65536);
  bn_relu_pool2<<<8192, 256, 0, stream>>>(RAW, MEAN, RSTD, g3, bb3, PAh, 128, 64, 64);

  // --- conv4 -> [16,32,32,256] -> bn+relu -> H4h ---
  conv_wmma<128,256,32,32,3,3,1><<<dim3(128, 8), 128, 0, stream>>>(PAh, WP4, cb4, RAW);
  bn_stats<<<256, 256, 0, stream>>>(RAW, MEAN, RSTD, 256, 16384);
  bn_relu<<<16384, 256, 0, stream>>>(RAW, MEAN, RSTD, g4, bb4, H4h, 256, 4194304);

  // --- proj conv -> [16,32,32,128] -> bn+relu -> P5h -> avgpool -> CAT ---
  conv_wmma<256,128,32,32,3,3,1><<<dim3(128, 4), 128, 0, stream>>>(H4h, WP5, pcb, RAW);
  bn_stats<<<128, 256, 0, stream>>>(RAW, MEAN, RSTD, 128, 16384);
  bn_relu<<<8192, 256, 0, stream>>>(RAW, MEAN, RSTD, pg, pbb, P5h, 128, 2097152);
  avgpool8_cat<<<128, 256, 0, stream>>>(P5h, CAT);

  // --- FFT power-spectrum branch ---
  fft256_rows<<<4096, 128, 0, stream>>>(x, FRE, FIM);
  fft256_cols<<<4096, 128, 0, stream>>>(FRE, FIM);
  power_radial<<<16, 32, 0, stream>>>(FRE, FIM, PF0);
  fc_wmma<true ><<<4, 128, 0, stream>>>(PF0, pw1, pb1, PF1,       32,  256,   32,  256);
  fc_wmma<true ><<<2, 128, 0, stream>>>(PF1, pw2, pb2, PF2,      256,  128,  256,  128);
  fc_wmma<true ><<<1, 128, 0, stream>>>(PF2, pw3, pb3, CAT+2048, 128,   64,  128, 2112);

  // --- head: [16,2112] -> 512 -> 256 -> 128 -> 2 ---
  fc_wmma<true ><<<8, 128, 0, stream>>>(CAT, fw1, fb1, FC1, 2112, 512, 2112, 512);
  fc_wmma<true ><<<4, 128, 0, stream>>>(FC1, fw2, fb2, FC2,  512, 256,  512, 256);
  fc_wmma<true ><<<2, 128, 0, stream>>>(FC2, fw3, fb3, FC3,  256, 128,  256, 128);
  fc_wmma<false><<<1, 128, 0, stream>>>(FC3, fw4, fb4, (float*)d_out, 128, 2, 128, 2);
}
